// MaternKernel_28149215658540
// MI455X (gfx1250) — matrix operand order repaired and checked by reading, not compiled, not run on hardware
//
#include <hip/hip_runtime.h>

// ---------------------------------------------------------------------------
// Matern-1.5 kernel matrix on MI455X (gfx1250, wave32, WMMA + TDM).
// bf16x3 split GEMM (hi/lo bf16 pairs, f32 accumulate) + fused Matern epilogue.
// Staging via Tensor Data Mover (tensor_load_to_lds) with double-buffered LDS.
// ---------------------------------------------------------------------------

typedef __bf16        v16bf  __attribute__((ext_vector_type(16)));
typedef float         v8f    __attribute__((ext_vector_type(8)));
typedef unsigned int  u32x4  __attribute__((ext_vector_type(4)));
typedef unsigned int  u32x4v __attribute__((ext_vector_type(4)));
typedef int           i32x8v __attribute__((ext_vector_type(8)));
typedef int           i32x4v __attribute__((ext_vector_type(4)));

#define DIM    256      // feature dimension (reference: d = 256)
#define BT     128      // block output tile (BT x BT)
#define KC     32       // K chunk per LDS stage (one WMMA K step)
#define LDSROW 40       // ushorts per LDS row: KC + 8 pad (80B -> conflict-free)
#define PANEL  (BT * LDSROW)          // one bf16 panel in LDS (ushorts)
#define LDS_BYTES (2 * 4 * PANEL * 2) // double-buffered 4 panels (81920 B)

#define SQRT3F 1.7320508075688772f

#if __has_builtin(__builtin_amdgcn_tensor_load_to_lds)
#define HAVE_TDM 1
#else
#define HAVE_TDM 0
#endif

__device__ __forceinline__ unsigned short f32_to_bf16(float f) {
  unsigned int u = __float_as_uint(f);
  u += 0x7FFFu + ((u >> 16) & 1u);          // round-to-nearest-even
  return (unsigned short)(u >> 16);
}
__device__ __forceinline__ float bf16_to_f32(unsigned short h) {
  return __uint_as_float(((unsigned int)h) << 16);
}

// ---------------------------------------------------------------------------
// Kernel 1: column mean of x1 and reciprocal lengthscale.
// ---------------------------------------------------------------------------
__global__ __launch_bounds__(256) void mean_kernel(
    const float* __restrict__ x1, const float* __restrict__ ls,
    float* __restrict__ mean, float* __restrict__ invls, int n1) {
  const int d = blockIdx.x;
  const int t = threadIdx.x;
  __shared__ float red[256];
  float s = 0.0f;
  for (int r = t; r < n1; r += 256) s += x1[(size_t)r * DIM + d];
  red[t] = s;
  __syncthreads();
#pragma unroll
  for (int off = 128; off > 0; off >>= 1) {
    if (t < off) red[t] += red[t + off];
    __syncthreads();
  }
  if (t == 0) {
    mean[d]  = red[0] / (float)n1;
    invls[d] = 1.0f / ls[d];
  }
}

// ---------------------------------------------------------------------------
// Kernel 2: y = (x - mean) * invls ; split into bf16 hi/lo ; row sq-norm (f32).
// ---------------------------------------------------------------------------
__global__ __launch_bounds__(256) void prep_kernel(
    const float* __restrict__ x, const float* __restrict__ mean,
    const float* __restrict__ invls,
    unsigned short* __restrict__ xh, unsigned short* __restrict__ xl,
    float* __restrict__ sq) {
  const int r = blockIdx.x;
  const int t = threadIdx.x;
  const size_t idx = (size_t)r * DIM + t;
  float y = (x[idx] - mean[t]) * invls[t];
  unsigned short h = f32_to_bf16(y);
  unsigned short l = f32_to_bf16(y - bf16_to_f32(h));
  xh[idx] = h;
  xl[idx] = l;
  __shared__ float red[256];
  red[t] = y * y;
  __syncthreads();
#pragma unroll
  for (int off = 128; off > 0; off >>= 1) {
    if (t < off) red[t] += red[t + off];
    __syncthreads();
  }
  if (t == 0) sq[r] = red[0];
}

// ---------------------------------------------------------------------------
// TDM issue: one 128x32-ushort tile (row stride 256) -> LDS with HW padding
// of 16B after every 64B row, reproducing the LDSROW=40 conflict-free layout.
// D# built per CDNA5 ISA. 6-arg builtin form
// (amdgpu-toolchain clang-23): (g0 4dw, g1 8dw, g2 4dw, g3 4dw, extra 8dw, cpol).
// ---------------------------------------------------------------------------
__device__ __forceinline__ void tdm_issue_panel(const unsigned short* src,
                                                const unsigned short* dstLds) {
#if HAVE_TDM
  const size_t ga = (size_t)src;
  u32x4v g0;
  g0[0] = 1u;                                        // count=1, user descriptor
  g0[1] = (unsigned int)(size_t)(const void*)dstLds; // lds_addr = flat[31:0]
  g0[2] = (unsigned int)(ga & 0xFFFFFFFFu);          // global_addr[31:0]
  g0[3] = (unsigned int)((ga >> 32) & 0x01FFFFFFu)   // global_addr[56:32]
          | (2u << 30);                              // type = 2 ("image")
  i32x8v g1;
  g1[0] = (1 << 16)      // data_size = 2 bytes
        | (1 << 20)      // pad_enable
        | (3 << 22)      // pad_interval: 16 DWORDs (one 64B tile row)
        | (3 << 25);     // pad_amount:   4 DWORDs (16B) -> 80B row stride
  g1[1] = (int)(DIM << 16);        // tensor_dim0 = 256 (atomic_barrier=0)
  g1[2] = (int)(8192 << 16);       // tensor_dim1 = 8192 rows (no OOB)
  g1[3] = (int)(KC << 16);         // tile_dim0 = 32 elements
  g1[4] = BT;                      // tile_dim1 = 128 rows, tile_dim2 = 0
  g1[5] = DIM;                     // tensor_dim0_stride = 256 elements
  g1[6] = 0;
  g1[7] = 0;
  i32x4v z4 = {0, 0, 0, 0};                    // groups 2/3 unused (2D tile)
  i32x8v z8 = {0, 0, 0, 0, 0, 0, 0, 0};        // extra descriptor words unused
  __builtin_amdgcn_tensor_load_to_lds(g0, g1, z4, z4, z8, 0);
#else
  (void)src; (void)dstLds;
#endif
}

__device__ __forceinline__ void wait_tensorcnt_4() {
#if __has_builtin(__builtin_amdgcn_s_wait_tensorcnt)
  __builtin_amdgcn_s_wait_tensorcnt(4);
#else
  asm volatile("s_wait_tensorcnt 0x4" ::: "memory");
#endif
}
__device__ __forceinline__ void wait_tensorcnt_0() {
#if __has_builtin(__builtin_amdgcn_s_wait_tensorcnt)
  __builtin_amdgcn_s_wait_tensorcnt(0);
#else
  asm volatile("s_wait_tensorcnt 0x0" ::: "memory");
#endif
}

// ---------------------------------------------------------------------------
// Fragment loaders per the CDNA5 WMMA VGPR layouts (CDNA5 ISA):
//   A 16x32 bf16: lane (g=ln>>4, lr=ln&15) row M=lr holds K=8g..8g+7 (V0-3)
//                 and K=16+8g..16+8g+7 (V4-7)  -> two b128 at +8g, +16+8g
//   B 32x16 bf16: lane col N=lr holds K=8g..8g+7 and K=16+8g..16+8g+7 -> two b128 at +8g, +16+8g
// ---------------------------------------------------------------------------
__device__ __forceinline__ v16bf load_a_frag(const unsigned short* base,
                                             int mrow, int g) {
  union { u32x4 q[2]; v16bf v; } f;
  const unsigned short* p = base + mrow * LDSROW + 8 * g;
  f.q[0] = *(const u32x4*)p;
  f.q[1] = *(const u32x4*)(p + 16);
  return f.v;
}
__device__ __forceinline__ v16bf load_b_frag(const unsigned short* base,
                                             int nrow, int g) {
  union { u32x4 q[2]; v16bf v; } f;
  const unsigned short* p = base + nrow * LDSROW + 8 * g;
  f.q[0] = *(const u32x4*)p;
  f.q[1] = *(const u32x4*)(p + 16);
  return f.v;
}

// ---------------------------------------------------------------------------
// Kernel 3: bf16x3 WMMA GEMM + Matern-1.5 epilogue.
// grid = (n2/BT, n1/BT), 256 threads = 8 waves; wave -> 32x64 sub-block
// (2 M-tiles x 4 N-tiles of 16x16). 8 K-chunks of 32, TDM double-buffered.
// ---------------------------------------------------------------------------
__global__ __launch_bounds__(256) void matern_gemm(
    const unsigned short* __restrict__ x1h, const unsigned short* __restrict__ x1l,
    const unsigned short* __restrict__ x2h, const unsigned short* __restrict__ x2l,
    const float* __restrict__ sq1, const float* __restrict__ sq2,
    float* __restrict__ out, int n2) {
  extern __shared__ unsigned short lds[];   // 2 buffers x 4 panels (81920 B)

  const int tid = threadIdx.x;
  const int M0  = blockIdx.y * BT;
  const int N0  = blockIdx.x * BT;

  const int wave = tid >> 5;     // 0..7
  const int ln   = tid & 31;
  const int g    = ln >> 4;      // lane half
  const int lr   = ln & 15;
  const int wm   = wave & 3;     // M group: rows 32*wm .. +31
  const int wn   = wave >> 2;    // N group: cols 64*wn .. +63

  const unsigned short* gsrc[4] = {
      x1h + (size_t)M0 * DIM, x1l + (size_t)M0 * DIM,
      x2h + (size_t)N0 * DIM, x2l + (size_t)N0 * DIM };

  // Scalar (wave-uniform) leader test: TDM issues regardless of EXEC, so the
  // guard must be a true scalar branch, not a lane mask.
  const bool leader = (__builtin_amdgcn_readfirstlane((int)tid) == 0);

  v8f c[2][4];
#pragma unroll
  for (int i = 0; i < 2; ++i)
#pragma unroll
    for (int j = 0; j < 4; ++j) c[i][j] = (v8f)(0.0f);

#if HAVE_TDM
  if (leader) {
#pragma unroll
    for (int a = 0; a < 4; ++a)
      tdm_issue_panel(gsrc[a], lds + a * PANEL);     // chunk 0 -> buf 0
  }
#endif

  int buf = 0;
  for (int kc = 0; kc < DIM; kc += KC) {
    unsigned short* base = lds + buf * (4 * PANEL);

#if HAVE_TDM
    if (leader) {
      if (kc + KC < DIM) {
        unsigned short* nbase = lds + (buf ^ 1) * (4 * PANEL);
#pragma unroll
        for (int a = 0; a < 4; ++a)
          tdm_issue_panel(gsrc[a] + (kc + KC), nbase + a * PANEL);
        wait_tensorcnt_4();   // current chunk's 4 DMAs complete (in-order)
      } else {
        wait_tensorcnt_0();
      }
    }
    __syncthreads();          // LDS data visible to all waves
#else
    __syncthreads();          // previous chunk's fragment reads complete
#pragma unroll
    for (int a = 0; a < 4; ++a) {
#pragma unroll
      for (int it = 0; it < 2; ++it) {
        int idx = tid + it * 256;
        int row = idx >> 2;
        int c4  = idx & 3;
        const u32x4* s = (const u32x4*)(gsrc[a] + (size_t)row * DIM + kc + c4 * 8);
        *(u32x4*)(base + a * PANEL + row * LDSROW + c4 * 8) = *s;
      }
    }
    __syncthreads();
#endif

    const unsigned short* Ah = base;
    const unsigned short* Al = Ah + PANEL;
    const unsigned short* Bh = Al + PANEL;
    const unsigned short* Bl = Bh + PANEL;

    v16bf ah[2], al[2], bh4[4], bl4[4];
#pragma unroll
    for (int i = 0; i < 2; ++i) {
      int mrow = 32 * wm + 16 * i + lr;
      ah[i] = load_a_frag(Ah, mrow, g);
      al[i] = load_a_frag(Al, mrow, g);
    }
#pragma unroll
    for (int j = 0; j < 4; ++j) {
      int nrow = 64 * wn + 16 * j + lr;
      bh4[j] = load_b_frag(Bh, nrow, g);
      bl4[j] = load_b_frag(Bl, nrow, g);
    }
#pragma unroll
    for (int i = 0; i < 2; ++i) {
#pragma unroll
      for (int j = 0; j < 4; ++j) {
        c[i][j] = __builtin_amdgcn_wmma_f32_16x16x32_bf16(
            false, ah[i], false, bh4[j], (short)0, c[i][j], false, false);
        c[i][j] = __builtin_amdgcn_wmma_f32_16x16x32_bf16(
            false, ah[i], false, bl4[j], (short)0, c[i][j], false, false);
        c[i][j] = __builtin_amdgcn_wmma_f32_16x16x32_bf16(
            false, al[i], false, bh4[j], (short)0, c[i][j], false, false);
      }
    }

#if HAVE_TDM
    __syncthreads();   // all fragment reads done before buf is DMA-overwritten
    buf ^= 1;
#endif
  }

  // Epilogue: d2 = |a|^2 + |b|^2 - 2*dot ; Matern nu=1.5.
  // C/D layout: VGPR r -> (M = r + 8g, N = lr) within each 16x16 tile.
#pragma unroll
  for (int i = 0; i < 2; ++i) {
    int rowb = M0 + 32 * wm + 16 * i + 8 * g;
#pragma unroll
    for (int j = 0; j < 4; ++j) {
      int colb = N0 + 64 * wn + 16 * j + lr;
      float q2 = sq2[colb];
#pragma unroll
      for (int r = 0; r < 8; ++r) {
        int row = rowb + r;
        float d2 = sq1[row] + q2 - 2.0f * c[i][j][r];
        d2 = fmaxf(d2, 0.0f);
        float dist = fmaxf(__builtin_sqrtf(d2), 1e-15f);
        float t = SQRT3F * dist;
        out[(size_t)row * (size_t)n2 + colb] = (1.0f + t) * __expf(-t);
      }
    }
  }
}

// ---------------------------------------------------------------------------
// Host launcher.
// ---------------------------------------------------------------------------
extern "C" void kernel_launch(void* const* d_in, const int* in_sizes, int n_in,
                              void* d_out, int out_size, void* d_ws, size_t ws_size,
                              hipStream_t stream) {
  (void)n_in; (void)out_size; (void)ws_size;
  const float* x1 = (const float*)d_in[0];
  const float* x2 = (const float*)d_in[1];
  const float* ls = (const float*)d_in[2];
  float* out = (float*)d_out;

  const int n1 = in_sizes[0] / DIM;   // 8192
  const int n2 = in_sizes[1] / DIM;   // 8192

  // Workspace layout (bytes): [mean 1K][invls 1K][sq1][sq2] ... bf16 panels
  char* ws = (char*)d_ws;
  float* mean  = (float*)(ws);
  float* invls = (float*)(ws + 1024);
  float* sq1   = (float*)(ws + 2048);
  float* sq2   = (float*)(ws + 2048 + 4 * (size_t)n1);
  unsigned short* x1h = (unsigned short*)(ws + 131072);
  unsigned short* x1l = x1h + (size_t)n1 * DIM;
  unsigned short* x2h = x1l + (size_t)n1 * DIM;
  unsigned short* x2l = x2h + (size_t)n2 * DIM;

  mean_kernel<<<DIM, 256, 0, stream>>>(x1, ls, mean, invls, n1);
  prep_kernel<<<n1, 256, 0, stream>>>(x1, mean, invls, x1h, x1l, sq1);
  prep_kernel<<<n2, 256, 0, stream>>>(x2, mean, invls, x2h, x2l, sq2);

  dim3 grid(n2 / BT, n1 / BT);
  matern_gemm<<<grid, 256, LDS_BYTES, stream>>>(x1h, x1l, x2h, x2l,
                                                sq1, sq2, out, n2);
}
